// TitanAttention_47158740910764
// MI455X (gfx1250) — matrix operand order repaired and checked by reading, not compiled, not run on hardware
//
#include <hip/hip_runtime.h>
#include <hip/hip_bf16.h>
#include <stdint.h>

typedef __bf16 bf16_t;
typedef __attribute__((ext_vector_type(16))) __bf16 v16bf;
typedef __attribute__((ext_vector_type(8)))  __bf16 v8bf;
typedef __attribute__((ext_vector_type(8)))  float   v8f;

union FragA { v16bf v; v8bf h[2]; };

__device__ __forceinline__ v8f wmma_bf16(v16bf a, v16bf b, v8f c) {
  // D = A(16x32) * B(32x16) + C, fp32 accumulate
  return __builtin_amdgcn_wmma_f32_16x16x32_bf16(false, a, false, b, (short)0, c, false, false);
}

// CDNA5 async memory->LDS (ASYNCcnt-tracked, no VGPR data path)
__device__ __forceinline__ void async_b128(uint32_t lds_off, const void* gptr) {
  asm volatile("global_load_async_to_lds_b128 %0, %1, off"
               :: "v"(lds_off), "v"(gptr) : "memory");
}
__device__ __forceinline__ void wait_async0() {
  asm volatile("s_wait_asynccnt 0" ::: "memory");
}

// lane-xor exchange via ds_swizzle (group-of-32 mode, immediate pattern):
// no VALU address math, unlike __shfl_xor's bpermute expansion.
template<int XM>
__device__ __forceinline__ float swz_xor(float v) {
  int i = __builtin_amdgcn_ds_swizzle(__float_as_int(v), (XM << 10) | 0x1f);
  return __int_as_float(i);
}

// ---------------------------------------------------------------------------
// elementwise cast f32 -> bf16
__global__ void cast_bf16_kernel(const float* __restrict__ in,
                                 bf16_t* __restrict__ out, int n) {
  int i = blockIdx.x * blockDim.x + threadIdx.x;
  if (i < n) out[i] = (bf16_t)in[i];
}

// repack weight to transposed bf16 [N][K] so B-fragments are contiguous
__global__ void pack_wT_kernel(const float* __restrict__ w,
                               bf16_t* __restrict__ out, int transpose) {
  int n = blockIdx.x;                       // 0..1023 (output col of GEMM)
  int k = blockIdx.y * 256 + threadIdx.x;   // 0..1023
  float v = transpose ? w[(size_t)k * 1024 + n] : w[(size_t)n * 1024 + k];
  out[(size_t)n * 1024 + k] = (bf16_t)v;
}

// ---------------------------------------------------------------------------
// GEMM: C[M][1024] = act(A[M][1024](bf16) * Bt[1024][1024]^T(bf16) + bias)
// 128 threads = 4 waves (2x2), wave tile 64x64, block tile 128x128,
// double-buffered async-to-LDS staging of A and B tiles.
#define GTM 128
#define GTN 128
#define GTK 64
#define LDA 80                          // bf16 elems per LDS row (160B)
#define TILEB (GTM * LDA * 2)           // 20480 B per matrix per stage

__device__ __forceinline__ void gemm_issue(
    const bf16_t* __restrict__ A, const bf16_t* __restrict__ Bt,
    uint32_t as_base, uint32_t bs_base, int buf, int m0, int n0, int k0,
    int srow, int schk)
{
  #pragma unroll
  for (int p = 0; p < 8; p++) {
    int row = srow + p * 16;
    const bf16_t* ga = A  + (size_t)(m0 + row) * 1024 + k0 + schk;
    const bf16_t* gb = Bt + (size_t)(n0 + row) * 1024 + k0 + schk;
    uint32_t lo = (uint32_t)(buf * TILEB + row * (LDA * 2) + schk * 2);
    async_b128(as_base + lo, ga);
    async_b128(bs_base + lo, gb);
  }
}

template<bool SILU>
__global__ __launch_bounds__(128) void gemm_bias_act(
    const bf16_t* __restrict__ A, const bf16_t* __restrict__ Bt,
    const float* __restrict__ bias, float* __restrict__ C)
{
  extern __shared__ __align__(32) char smem[];
  bf16_t* As = (bf16_t*)smem;                    // [2][GTM*LDA]
  bf16_t* Bs = (bf16_t*)(smem + 2 * TILEB);      // [2][GTN*LDA]
  const uint32_t as_base = (uint32_t)(uintptr_t)As;   // LDS byte offset
  const uint32_t bs_base = (uint32_t)(uintptr_t)Bs;

  const int tid = threadIdx.x, lane = tid & 31, wave = tid >> 5;
  const int wm = (wave >> 1) * 64, wn = (wave & 1) * 64;
  const int m0 = blockIdx.y * GTM, n0 = blockIdx.x * GTN;

  v8f acc[4][4];
  #pragma unroll
  for (int i = 0; i < 4; i++)
    #pragma unroll
    for (int j = 0; j < 4; j++)
      #pragma unroll
      for (int e = 0; e < 8; e++) acc[i][j][e] = 0.f;

  const int srow = tid >> 3;          // 0..15
  const int schk = (tid & 7) * 8;     // bf16 k-offset, 16B chunks

  gemm_issue(A, Bt, as_base, bs_base, 0, m0, n0, 0, srow, schk);
  wait_async0();
  __syncthreads();

  #pragma unroll 2
  for (int k0 = 0; k0 < 1024; k0 += GTK) {
    const int buf = (k0 >> 6) & 1;
    const bool more = (k0 + GTK) < 1024;
    if (more)
      gemm_issue(A, Bt, as_base, bs_base, buf ^ 1, m0, n0, k0 + GTK, srow, schk);

    const bf16_t* Ac = As + buf * (GTM * LDA);
    const bf16_t* Bc = Bs + buf * (GTN * LDA);
    #pragma unroll
    for (int ks = 0; ks < GTK; ks += 32) {
      // hoist all fragments first: 16 independent ds_load_b128 in flight
      FragA a[4], bfr[4];
      #pragma unroll
      for (int i = 0; i < 4; i++) {
        int row = wm + i * 16 + (lane & 15);
        int kb  = ks + ((lane >> 4) << 4);       // lanes>=16: K+16
        a[i].h[0] = *(const v8bf*)&Ac[row * LDA + kb];
        a[i].h[1] = *(const v8bf*)&Ac[row * LDA + kb + 8];
      }
      #pragma unroll
      for (int j = 0; j < 4; j++) {
        int n  = wn + j * 16 + (lane & 15);
        int kb = ks + ((lane >> 4) << 4);        // lanes>=16: K+16
        bfr[j].v = *(const v16bf*)&Bc[n * LDA + kb];
      }
      #pragma unroll
      for (int j = 0; j < 4; j++)
        #pragma unroll
        for (int i = 0; i < 4; i++)
          acc[i][j] = wmma_bf16(a[i].v, bfr[j].v, acc[i][j]);
    }
    if (more) { wait_async0(); __syncthreads(); }
  }

  // epilogue: C layout -> VGPR r: lanes 0-15 row r, lanes 16-31 row r+8
  const int rl = (lane >> 4) << 3;
  const int cl = lane & 15;
  #pragma unroll
  for (int i = 0; i < 4; i++) {
    int row = m0 + wm + i * 16 + rl;
    #pragma unroll
    for (int j = 0; j < 4; j++) {
      int col = n0 + wn + j * 16 + cl;
      float bv = bias[col];
      #pragma unroll
      for (int r = 0; r < 8; r++) {
        float xv = acc[i][j][r] + bv;
        if (SILU) xv = xv / (1.f + __expf(-xv));
        C[(size_t)(row + r) * 1024 + col] = xv;
      }
    }
  }
}

// ---------------------------------------------------------------------------
// depthwise conv k=3 pad=1 over seq + bias, f32 in -> bf16 out
__global__ __launch_bounds__(256) void dsconv_kernel(
    const float* __restrict__ in, const float* __restrict__ dw,
    const float* __restrict__ dwb, bf16_t* __restrict__ out)
{
  int row = blockIdx.x;        // b*2048 + s
  int s = row & 2047;
  #pragma unroll
  for (int i = 0; i < 4; i++) {
    int c = threadIdx.x + i * 256;
    float w0 = dw[c * 3 + 0], w1 = dw[c * 3 + 1], w2 = dw[c * 3 + 2];
    const float* base = in + (size_t)row * 1024 + c;
    float acc = dwb[c] + base[0] * w1;
    if (s > 0)    acc += base[-1024] * w0;
    if (s < 2047) acc += base[ 1024] * w2;
    out[(size_t)row * 1024 + c] = (bf16_t)acc;
  }
}

// ---------------------------------------------------------------------------
// L2-normalize over 1024 dims, scale, write head-major bf16 [bh][s][64]
__global__ __launch_bounds__(256) void l2norm_heads_kernel(
    const float* __restrict__ in, bf16_t* __restrict__ out, float scale)
{
  __shared__ float red[8];
  int row = blockIdx.x;                 // b*2048 + s
  int b = row >> 11, s = row & 2047;
  float v[4]; float ss = 0.f;
  #pragma unroll
  for (int i = 0; i < 4; i++) {
    v[i] = in[(size_t)row * 1024 + threadIdx.x + i * 256];
    ss += v[i] * v[i];
  }
  ss += swz_xor<16>(ss);
  ss += swz_xor<8>(ss);
  ss += swz_xor<4>(ss);
  ss += swz_xor<2>(ss);
  ss += swz_xor<1>(ss);
  if ((threadIdx.x & 31) == 0) red[threadIdx.x >> 5] = ss;
  __syncthreads();
  float tot = 0.f;
  #pragma unroll
  for (int w = 0; w < 8; w++) tot += red[w];
  float inv = scale / fmaxf(sqrtf(tot), 1e-12f);
  #pragma unroll
  for (int i = 0; i < 4; i++) {
    int c = threadIdx.x + i * 256;
    int h = c >> 6, d = c & 63;
    out[(((size_t)b * 16 + h) * 2048 + s) * 64 + d] = (bf16_t)(v[i] * inv);
  }
}

// V -> bf16, transposed to d-major [bh][64][2048]
__global__ __launch_bounds__(256) void vtrans_kernel(
    const float* __restrict__ in, bf16_t* __restrict__ out)
{
  int row = blockIdx.x;
  int b = row >> 11, s = row & 2047;
  #pragma unroll
  for (int i = 0; i < 4; i++) {
    int c = threadIdx.x + i * 256;
    int h = c >> 6, d = c & 63;
    out[(((size_t)b * 16 + h) * 64 + d) * 2048 + s] =
        (bf16_t)in[(size_t)row * 1024 + c];
  }
}

// ---------------------------------------------------------------------------
// Flash-style attention. Block = (b,h, 128 q-rows), 8 waves x 16 q-rows.
// Double-buffered async-to-LDS staging of 64-key K/V tiles.
#define LDK 80
#define KTILEB (64 * LDK * 2)     // 10240 B per K (or V) tile stage

__global__ __launch_bounds__(256) void attention_kernel(
    const bf16_t* __restrict__ Qh, const bf16_t* __restrict__ Kh,
    const bf16_t* __restrict__ Vt, const int* __restrict__ mask,
    bf16_t* __restrict__ AO)
{
  extern __shared__ __align__(32) char smem[];
  bf16_t* Ks = (bf16_t*)smem;                        // [2][64*LDK]
  bf16_t* Vs = (bf16_t*)(smem + 2 * KTILEB);         // [2][64*LDK]
  bf16_t* Ps = (bf16_t*)(smem + 4 * KTILEB);         // [8][16*LDK]
  const uint32_t ks_base = (uint32_t)(uintptr_t)Ks;
  const uint32_t vs_base = (uint32_t)(uintptr_t)Vs;

  const int tid = threadIdx.x, lane = tid & 31, wave = tid >> 5;
  const int bh = blockIdx.y, b = bh >> 4, h = bh & 15;
  const int q0 = blockIdx.x * 128 + wave * 16;

  // Q fragments (d = 0..31 and 32..63), resident for the whole kernel
  FragA qf[2];
  {
    const bf16_t* Qb = Qh + ((size_t)bh * 2048 + q0) * 64;
    int r = lane & 15, kb = (lane >> 4) << 4;
    #pragma unroll
    for (int dh = 0; dh < 2; dh++) {
      qf[dh].h[0] = *(const v8bf*)(Qb + r * 64 + dh * 32 + kb);
      qf[dh].h[1] = *(const v8bf*)(Qb + r * 64 + dh * 32 + kb + 8);
    }
  }

  v8f O[4];
  float mrow[8], lrow[8];
  #pragma unroll
  for (int dt = 0; dt < 4; dt++)
    #pragma unroll
    for (int e = 0; e < 8; e++) O[dt][e] = 0.f;
  #pragma unroll
  for (int r = 0; r < 8; r++) { mrow[r] = -3.0e38f; lrow[r] = 0.f; }

  const int ldr  = tid >> 2;             // 0..63
  const int lofs = (tid & 3) * 16;       // bf16 chunk offset

  // issue K/V tile (jb) into stage buf
  auto issue = [&](int buf, int jb) {
    const bf16_t* gk = Kh + ((size_t)bh * 2048 + jb + ldr) * 64 + lofs;
    const bf16_t* gv = Vt + ((size_t)bh * 64 + ldr) * 2048 + jb + lofs;
    uint32_t lo = (uint32_t)(buf * KTILEB + ldr * (LDK * 2) + lofs * 2);
    async_b128(ks_base + lo,      gk);
    async_b128(ks_base + lo + 16, gk + 8);
    async_b128(vs_base + lo,      gv);
    async_b128(vs_base + lo + 16, gv + 8);
  };

  issue(0, 0);
  wait_async0();
  __syncthreads();

  for (int jb = 0; jb < 2048; jb += 64) {
    const int buf = (jb >> 6) & 1;
    const bool more = (jb + 64) < 2048;
    if (more) issue(buf ^ 1, jb + 64);

    const bf16_t* Kc = Ks + buf * (64 * LDK);
    const bf16_t* Vc = Vs + buf * (64 * LDK);

    // branchless mask term per score column (loaded early, hidden under wmma)
    float ma[4];
    #pragma unroll
    for (int nt = 0; nt < 4; nt++) {
      int mk = mask[b * 2048 + jb + nt * 16 + (lane & 15)];
      ma[nt] = mk ? 0.f : -1.0e30f;
    }

    // scores: 16 q x 64 k (q pre-scaled by 1/sqrt(64)); hoist all 8 K-frags
    FragA bk[4][2];
    #pragma unroll
    for (int nt = 0; nt < 4; nt++) {
      int n = nt * 16 + (lane & 15);
      #pragma unroll
      for (int dh = 0; dh < 2; dh++)
        bk[nt][dh].v =
            *(const v16bf*)&Kc[n * LDK + dh * 32 + ((lane >> 4) << 4)];
    }
    v8f sc[4];
    #pragma unroll
    for (int nt = 0; nt < 4; nt++) {
      v8f s;
      #pragma unroll
      for (int e = 0; e < 8; e++) s[e] = 0.f;
      s = wmma_bf16(qf[0].v, bk[nt][0].v, s);
      s = wmma_bf16(qf[1].v, bk[nt][1].v, s);
      #pragma unroll
      for (int r = 0; r < 8; r++) s[r] += ma[nt];
      sc[nt] = s;
    }

    // online softmax (base-2); reductions over the 16-lane half via ds_swizzle
    const float LOG2E = 1.44269504088896f;
    #pragma unroll
    for (int r = 0; r < 8; r++) {
      float mx = fmaxf(fmaxf(sc[0][r], sc[1][r]), fmaxf(sc[2][r], sc[3][r]));
      mx = fmaxf(mx, swz_xor<1>(mx));
      mx = fmaxf(mx, swz_xor<2>(mx));
      mx = fmaxf(mx, swz_xor<4>(mx));
      mx = fmaxf(mx, swz_xor<8>(mx));
      float mnew  = fmaxf(mrow[r], mx);
      float alpha = exp2f((mrow[r] - mnew) * LOG2E);
      mrow[r] = mnew;
      float s0 = 0.f;
      #pragma unroll
      for (int nt = 0; nt < 4; nt++) {
        float p = exp2f((sc[nt][r] - mnew) * LOG2E);
        sc[nt][r] = p; s0 += p;
      }
      s0 += swz_xor<1>(s0);
      s0 += swz_xor<2>(s0);
      s0 += swz_xor<4>(s0);
      s0 += swz_xor<8>(s0);
      lrow[r] = lrow[r] * alpha + s0;
      #pragma unroll
      for (int dt = 0; dt < 4; dt++) O[dt][r] *= alpha;
    }
    // P (C-layout) -> per-wave LDS -> A-fragment layout
    {
      int rb = (lane >> 4) << 3, cb = lane & 15;
      bf16_t* pw = &Ps[wave * 16 * LDK];
      #pragma unroll
      for (int nt = 0; nt < 4; nt++)
        #pragma unroll
        for (int r = 0; r < 8; r++)
          pw[(rb + r) * LDK + nt * 16 + cb] = (bf16_t)sc[nt][r];
    }
    asm volatile("s_wait_dscnt 0" ::: "memory");
    FragA pf[2];
    {
      int r = lane & 15, kb = (lane >> 4) << 4;
      const bf16_t* pw = &Ps[wave * 16 * LDK];
      #pragma unroll
      for (int kh = 0; kh < 2; kh++) {
        pf[kh].h[0] = *(const v8bf*)(pw + r * LDK + kh * 32 + kb);
        pf[kh].h[1] = *(const v8bf*)(pw + r * LDK + kh * 32 + kb + 8);
      }
    }
    // O += P * V ; hoist all 8 V-frags first
    FragA bv[4][2];
    #pragma unroll
    for (int dt = 0; dt < 4; dt++)
      #pragma unroll
      for (int kh = 0; kh < 2; kh++)
        bv[dt][kh].v = *(const v16bf*)
            &Vc[(dt * 16 + (lane & 15)) * LDK + kh * 32 + ((lane >> 4) << 4)];
    #pragma unroll
    for (int dt = 0; dt < 4; dt++)
      #pragma unroll
      for (int kh = 0; kh < 2; kh++)
        O[dt] = wmma_bf16(pf[kh].v, bv[dt][kh].v, O[dt]);

    if (more) { wait_async0(); __syncthreads(); }
  }

  // finalize: write bf16 A-matrix for the output projection, [b,s, h*64+d]
  int rb = (lane >> 4) << 3, cb = lane & 15;
  #pragma unroll
  for (int r = 0; r < 8; r++) {
    float inv = 1.f / fmaxf(lrow[r], 1e-30f);
    int srow = q0 + rb + r;
    size_t base = ((size_t)b * 2048 + srow) * 1024 + h * 64;
    #pragma unroll
    for (int dt = 0; dt < 4; dt++)
      AO[base + dt * 16 + cb] = (bf16_t)(O[dt][r] * inv);
  }
}

// ---------------------------------------------------------------------------
extern "C" void kernel_launch(void* const* d_in, const int* in_sizes, int n_in,
                              void* d_out, int out_size, void* d_ws, size_t ws_size,
                              hipStream_t stream) {
  (void)in_sizes; (void)n_in; (void)out_size; (void)ws_size;
  const float* x    = (const float*)d_in[0];
  const int*   mask = (const int*)d_in[1];
  const float* W[3]   = { (const float*)d_in[2],  (const float*)d_in[8],  (const float*)d_in[14] };
  const float* Bv[3]  = { (const float*)d_in[3],  (const float*)d_in[9],  (const float*)d_in[15] };
  const float* DW[3]  = { (const float*)d_in[4],  (const float*)d_in[10], (const float*)d_in[16] };
  const float* DWB[3] = { (const float*)d_in[5],  (const float*)d_in[11], (const float*)d_in[17] };
  const float* PW[3]  = { (const float*)d_in[6],  (const float*)d_in[12], (const float*)d_in[18] };
  const float* PWB[3] = { (const float*)d_in[7],  (const float*)d_in[13], (const float*)d_in[19] };
  const float* wo = (const float*)d_in[20];
  const float* bo = (const float*)d_in[21];
  float* out = (float*)d_out;

  const size_t M = 8192;                 // B*S
  char* ws = (char*)d_ws;
  size_t off = 0;
  bf16_t* xb = (bf16_t*)(ws + off); off += M * 1024 * 2;          // 16 MB
  bf16_t* wT[7];
  for (int i = 0; i < 7; i++) { wT[i] = (bf16_t*)(ws + off); off += (size_t)1024 * 1024 * 2; }
  float*  bufA = (float*)(ws + off);  off += M * 1024 * 4;        // 32 MB
  bf16_t* bufB = (bf16_t*)(ws + off); off += M * 1024 * 2;        // 16 MB
  bf16_t* Qh   = (bf16_t*)(ws + off); off += M * 1024 * 2;        // 16 MB
  bf16_t* Kh   = (bf16_t*)(ws + off); off += M * 1024 * 2;        // 16 MB
  bf16_t* Vt   = (bf16_t*)(ws + off); off += M * 1024 * 2;        // 16 MB
  bf16_t* AO   = (bf16_t*)(ws + off); off += M * 1024 * 2;        // 16 MB

  const dim3 gemm_blk(128);
  const dim3 gemm_grid(1024 / GTN, M / GTM);
  const size_t gemm_lds = 4 * (size_t)TILEB;        // 80 KB (double-buffered A+B)
  const dim3 blk(256);
  const size_t att_lds = 4 * (size_t)KTILEB + 8 * 16 * LDK * 2;   // 60 KB

  // stage 0: casts / weight repacks
  cast_bf16_kernel<<<(int)(M * 1024 / 256), blk, 0, stream>>>(x, xb, (int)(M * 1024));
  dim3 pgrid(1024, 4);
  for (int p = 0; p < 3; p++) {
    pack_wT_kernel<<<pgrid, blk, 0, stream>>>(W[p],  wT[p],     1); // [in,out] -> T
    pack_wT_kernel<<<pgrid, blk, 0, stream>>>(PW[p], wT[3 + p], 0); // [out,in] -> direct
  }
  pack_wT_kernel<<<pgrid, blk, 0, stream>>>(wo, wT[6], 1);

  // stage 1: per-projection pipeline
  for (int p = 0; p < 3; p++) {
    gemm_bias_act<true><<<gemm_grid, gemm_blk, gemm_lds, stream>>>(xb, wT[p], Bv[p], bufA);
    dsconv_kernel<<<(int)M, blk, 0, stream>>>(bufA, DW[p], DWB[p], bufB);
    gemm_bias_act<false><<<gemm_grid, gemm_blk, gemm_lds, stream>>>(bufB, wT[3 + p], PWB[p], bufA);
    if (p == 0)      l2norm_heads_kernel<<<(int)M, blk, 0, stream>>>(bufA, Qh, 0.125f);
    else if (p == 1) l2norm_heads_kernel<<<(int)M, blk, 0, stream>>>(bufA, Kh, 1.0f);
    else             vtrans_kernel<<<(int)M, blk, 0, stream>>>(bufA, Vt);
  }

  // stage 2: attention
  attention_kernel<<<dim3(2048 / 128, 64), blk, att_lds, stream>>>(Qh, Kh, Vt, mask, AO);

  // stage 3: output projection
  gemm_bias_act<false><<<gemm_grid, gemm_blk, gemm_lds, stream>>>(AO, wT[6], bo, out);
}
